// BiMambaBlock_67027259621405
// MI455X (gfx1250) — compile-verified, matrix operand order checked by reading, not run on hardware
//
#include <hip/hip_runtime.h>
#include <hip/hip_bf16.h>
#include <math.h>

#define BATCH   4
#define SEQLEN  4096
#define DM      1024          // d_model
#define DI      2048          // d_inner
#define DTR     64            // dt_rank
#define DS      16            // d_state
#define NTOK    (BATCH * SEQLEN)   // 16384
#define XPROJ_N (DTR + 2 * DS)     // 96
#define SPAD    40            // padded LDS tile row stride (bf16 elems)

typedef __attribute__((ext_vector_type(8)))  __bf16 v8bf;
typedef __attribute__((ext_vector_type(16))) __bf16 v16bf;
typedef __attribute__((ext_vector_type(8)))  float  v8f;

// ---------------------------------------------------------------------------
// Async global->LDS copy of 32 bytes (two b128 transfers), ASYNCcnt-tracked.
// Builtin prototype (revealed by the round-2 diagnostic): takes global/local
// pointers to 16-byte int vectors plus immediate offset + cpol.
// ---------------------------------------------------------------------------
typedef int v4i_a __attribute__((vector_size(16)));
typedef __attribute__((address_space(1))) v4i_a* gas_t;
typedef __attribute__((address_space(3))) v4i_a* las_t;

__device__ __forceinline__ void async_copy32(const __bf16* g, __bf16* l) {
#if __has_builtin(__builtin_amdgcn_global_load_async_to_lds_b128)
    gas_t gp = (gas_t)(uintptr_t)g;
    las_t lp = (las_t)(uintptr_t)(uint32_t)(uintptr_t)l;  // low 32 bits of flat = LDS offset
    __builtin_amdgcn_global_load_async_to_lds_b128(gp,     lp,     0, 0);
    __builtin_amdgcn_global_load_async_to_lds_b128(gp + 1, lp + 1, 0, 0);
#else
    uint64_t ga = (uint64_t)(uintptr_t)g;
    uint32_t la = (uint32_t)(uintptr_t)l;
    asm volatile("global_load_async_to_lds_b128 %0, %1, off"
                 :: "v"(la), "v"(ga) : "memory");
    asm volatile("global_load_async_to_lds_b128 %0, %1, off offset:16"
                 :: "v"(la), "v"(ga) : "memory");
#endif
}

__device__ __forceinline__ void wait_async0() {
#if __has_builtin(__builtin_amdgcn_s_wait_asynccnt)
    __builtin_amdgcn_s_wait_asynccnt(0);
#else
    asm volatile("s_wait_asynccnt 0x0" ::: "memory");
#endif
}

// ---------------------------------------------------------------------------
// WMMA fragment layout (ISA 7.12.2, 16-bit A 16x32): lane<16 holds row M=lane,
// K = {kb..kb+7, kb+16..kb+23}, kb = (lane>>4)*8; lanes>=16 the +8/+24 halves.
// B (NxK row-major weight) uses the same pattern with N in place of M.
// ---------------------------------------------------------------------------
__device__ __forceinline__ v16bf frag_from(const __bf16* p) {
    v8bf lo = *(const v8bf*)(p);
    v8bf hi = *(const v8bf*)(p + 16);
    v16bf r;
#pragma unroll
    for (int i = 0; i < 8; ++i) { r[i] = lo[i]; r[i + 8] = hi[i]; }
    return r;
}

__device__ __forceinline__ v16bf load_frag_bf16(const __bf16* __restrict__ base,
                                                int ld, int row, int kb) {
    return frag_from(base + (size_t)row * (size_t)ld + kb);
}

// ---------------------------------------------------------------------------
// LDS-staged GEMM: C(MxN,f32) = A(MxK,bf16) * B(NxK,bf16)^T.
// block = 128 threads (4 waves) -> 64x64 block tile, 32x32 per wave.
// Tiles staged to LDS with async-to-LDS b128, double buffered.
// grid = (N/64, M/64). Requires M%64==0, N%64==0, K%32==0.
// ---------------------------------------------------------------------------
__global__ void gemm_bf16_wmma_lds(const __bf16* __restrict__ A,
                                   const __bf16* __restrict__ Bw,
                                   float* __restrict__ C,
                                   int M, int N, int K) {
    __shared__ __bf16 sA[2][64 * SPAD];
    __shared__ __bf16 sB[2][64 * SPAD];

    const int t    = threadIdx.x;
    const int lane = t & 31;
    const int w    = t >> 5;
    const int half = lane >> 4;
    const int rr   = lane & 15;
    const int wm   = (w >> 1) * 32;          // wave row offset in block tile
    const int wn   = (w & 1) * 32;           // wave col offset in block tile
    const int bm   = blockIdx.y * 64;
    const int bn   = blockIdx.x * 64;

    // staging: thread t copies 16 elems (32B) of row (t>>1), segment (t&1)*16
    const int sr   = t >> 1;
    const int sseg = (t & 1) * 16;

    v8f acc00 = {}, acc01 = {}, acc10 = {}, acc11 = {};
    const int nk = K / 32;

    // prologue: stage K-step 0 into buffer 0
    async_copy32(A  + (size_t)(bm + sr) * K + sseg, &sA[0][sr * SPAD + sseg]);
    async_copy32(Bw + (size_t)(bn + sr) * K + sseg, &sB[0][sr * SPAD + sseg]);

    for (int i = 0; i < nk; ++i) {
        wait_async0();
        __syncthreads();
        const int cur = i & 1;
        if (i + 1 < nk) {
            const int kn = (i + 1) * 32;
            async_copy32(A  + (size_t)(bm + sr) * K + kn + sseg, &sA[cur ^ 1][sr * SPAD + sseg]);
            async_copy32(Bw + (size_t)(bn + sr) * K + kn + sseg, &sB[cur ^ 1][sr * SPAD + sseg]);
        }
        const int kb = half * 8;
        v16bf a0 = frag_from(&sA[cur][(wm + rr) * SPAD + kb]);
        v16bf a1 = frag_from(&sA[cur][(wm + 16 + rr) * SPAD + kb]);
        v16bf b0 = frag_from(&sB[cur][(wn + rr) * SPAD + kb]);
        v16bf b1 = frag_from(&sB[cur][(wn + 16 + rr) * SPAD + kb]);
        acc00 = __builtin_amdgcn_wmma_f32_16x16x32_bf16(false, a0, false, b0, (short)0, acc00, false, false);
        acc01 = __builtin_amdgcn_wmma_f32_16x16x32_bf16(false, a0, false, b1, (short)0, acc01, false, false);
        acc10 = __builtin_amdgcn_wmma_f32_16x16x32_bf16(false, a1, false, b0, (short)0, acc10, false, false);
        acc11 = __builtin_amdgcn_wmma_f32_16x16x32_bf16(false, a1, false, b1, (short)0, acc11, false, false);
    }

    const int m0 = bm + wm, n0 = bn + wn;
#pragma unroll
    for (int g = 0; g < 8; ++g) {
        const size_t row0 = (size_t)(m0 + g + 8 * half) * N;
        const size_t row1 = (size_t)(m0 + 16 + g + 8 * half) * N;
        C[row0 + n0 + rr]      = acc00[g];
        C[row0 + n0 + 16 + rr] = acc01[g];
        C[row1 + n0 + rr]      = acc10[g];
        C[row1 + n0 + 16 + rr] = acc11[g];
    }
}

// ---------------------------------------------------------------------------
// Direct-from-global GEMM (for N granularity 32; used by x_proj, N=96).
// One wave per 32x32 tile; grid = (N/32, M/32), block = 32.
// ---------------------------------------------------------------------------
__global__ void gemm_bf16_wmma(const __bf16* __restrict__ A,
                               const __bf16* __restrict__ Bw,
                               float* __restrict__ C,
                               int M, int N, int K) {
    const int lane = threadIdx.x & 31;
    const int half = lane >> 4;
    const int r    = lane & 15;
    const int n0   = blockIdx.x * 32;
    const int m0   = blockIdx.y * 32;

    v8f acc00 = {}, acc01 = {}, acc10 = {}, acc11 = {};

    for (int k0 = 0; k0 < K; k0 += 32) {
        const int kb = k0 + half * 8;
        v16bf a0 = load_frag_bf16(A,  K, m0 + r,      kb);
        v16bf a1 = load_frag_bf16(A,  K, m0 + 16 + r, kb);
        v16bf b0 = load_frag_bf16(Bw, K, n0 + r,      kb);
        v16bf b1 = load_frag_bf16(Bw, K, n0 + 16 + r, kb);
        acc00 = __builtin_amdgcn_wmma_f32_16x16x32_bf16(false, a0, false, b0, (short)0, acc00, false, false);
        acc01 = __builtin_amdgcn_wmma_f32_16x16x32_bf16(false, a0, false, b1, (short)0, acc01, false, false);
        acc10 = __builtin_amdgcn_wmma_f32_16x16x32_bf16(false, a1, false, b0, (short)0, acc10, false, false);
        acc11 = __builtin_amdgcn_wmma_f32_16x16x32_bf16(false, a1, false, b1, (short)0, acc11, false, false);
    }

#pragma unroll
    for (int g = 0; g < 8; ++g) {
        const size_t row0 = (size_t)(m0 + g + 8 * half) * N;
        const size_t row1 = (size_t)(m0 + 16 + g + 8 * half) * N;
        C[row0 + n0 + r]      = acc00[g];
        C[row0 + n0 + 16 + r] = acc01[g];
        C[row1 + n0 + r]      = acc10[g];
        C[row1 + n0 + 16 + r] = acc11[g];
    }
}

// ---------------------------------------------------------------------------
// f32 -> bf16 conversion (grid-stride)
// ---------------------------------------------------------------------------
__global__ void cvt_f32_bf16(const float* __restrict__ in,
                             __bf16* __restrict__ out, size_t n) {
    for (size_t i = (size_t)blockIdx.x * blockDim.x + threadIdx.x; i < n;
         i += (size_t)gridDim.x * blockDim.x)
        out[i] = (__bf16)in[i];
}

// extract dt_low (first DTR columns of x_dbl, row stride XPROJ_N) as bf16
__global__ void extract_dt_low(const float* __restrict__ x_dbl,
                               __bf16* __restrict__ out, size_t n) {
    for (size_t i = (size_t)blockIdx.x * blockDim.x + threadIdx.x; i < n;
         i += (size_t)gridDim.x * blockDim.x) {
        size_t row = i / DTR;
        int    col = (int)(i % DTR);
        out[i] = (__bf16)x_dbl[row * XPROJ_N + col];
    }
}

// ---------------------------------------------------------------------------
// causal depthwise conv1d (window 4, left pad 3) + bias + SiLU.
// x lives in the first DI columns of the xz buffer (row stride 2*DI).
// ---------------------------------------------------------------------------
__global__ void conv_silu(const float* __restrict__ xz,
                          const float* __restrict__ conv_w,
                          const float* __restrict__ conv_b,
                          float* __restrict__ xc_f32,
                          __bf16* __restrict__ xc_bf16) {
    size_t i = (size_t)blockIdx.x * blockDim.x + threadIdx.x;
    if (i >= (size_t)NTOK * DI) return;
    const int    d  = (int)(i % DI);
    const size_t bl = i / DI;              // b*SEQLEN + l
    const int    l  = (int)(bl % SEQLEN);
    const float* w  = conv_w + (size_t)d * 4;

    float acc = conv_b[d];
#pragma unroll
    for (int j = 0; j < 4; ++j) {
        int ls = l - 3 + j;
        if (ls >= 0)
            acc += w[j] * xz[(bl - (size_t)(l - ls)) * (2 * DI) + d];
    }
    float s = acc / (1.f + __expf(-acc));   // SiLU
    xc_f32[i]  = s;
    xc_bf16[i] = (__bf16)s;
}

// dt = softplus(dtmm + dt_proj_b), in place
__global__ void dt_softplus(float* __restrict__ dtmm,
                            const float* __restrict__ dt_b, size_t n) {
    for (size_t i = (size_t)blockIdx.x * blockDim.x + threadIdx.x; i < n;
         i += (size_t)gridDim.x * blockDim.x) {
        int   d = (int)(i % DI);
        float v = dtmm[i] + dt_b[d];
        dtmm[i] = (v > 20.f) ? v : log1pf(__expf(v));
    }
}

// ---------------------------------------------------------------------------
// Selective scan. One lane per (b,d) channel; h[16] in VGPRs; sequential L
// (the per-step clip on h forbids an associative/chunked scan).
// Fuses  y += x*D  and  y *= silu(z)  and writes y as bf16 for out_proj.
// ---------------------------------------------------------------------------
__global__ void selective_scan(const float* __restrict__ dt,
                               const float* __restrict__ xc,
                               const float* __restrict__ xz,     // z in cols [DI,2DI)
                               const float* __restrict__ x_dbl,  // B at col 64, C at 80
                               const float* __restrict__ A_log,
                               const float* __restrict__ Dp,
                               __bf16* __restrict__ y_bf16) {
    const int ch = blockIdx.x * blockDim.x + threadIdx.x;  // 0..BATCH*DI-1
    const int b  = ch >> 11;      // / DI
    const int d  = ch & (DI - 1);

    float A[DS], h[DS];
#pragma unroll
    for (int n = 0; n < DS; ++n) {
        A[n] = -__expf(A_log[(size_t)d * DS + n]);
        h[n] = 0.f;
    }
    const float  Dd   = Dp[d];
    const size_t row0 = (size_t)b * SEQLEN;

    for (int t = 0; t < SEQLEN; ++t) {
        const size_t rr = row0 + t;
        const float dtv = dt[rr * DI + d];
        const float xv  = xc[rr * DI + d];
        const float zv  = xz[rr * (2 * DI) + DI + d];
        const float4* Bp = (const float4*)(x_dbl + rr * XPROJ_N + DTR);
        const float4* Cp = (const float4*)(x_dbl + rr * XPROJ_N + DTR + DS);
        float4 Bv[4] = {Bp[0], Bp[1], Bp[2], Bp[3]};
        float4 Cv[4] = {Cp[0], Cp[1], Cp[2], Cp[3]};
        if (t + 1 < SEQLEN)   // global_prefetch for next step's B/C row
            __builtin_prefetch(x_dbl + (rr + 1) * XPROJ_N + DTR, 0, 1);

        const float dtc = fminf(fmaxf(dtv, -10.f), 10.f);
        const float xcl = fminf(fmaxf(xv,  -10.f), 10.f);

        float S = 0.f;
#pragma unroll
        for (int n = 0; n < DS; ++n) {
            float e = fminf(fmaxf(dtc * A[n], -50.f), 50.f);
            S += __expf(e);
        }
        const float dtx = dtc * xcl;
        const float* Bf = (const float*)Bv;
        const float* Cf = (const float*)Cv;
        float y = 0.f;
#pragma unroll
        for (int n = 0; n < DS; ++n) {
            float hn = fminf(fmaxf(S * h[n] + dtx * Bf[n], -100.f), 100.f);
            h[n] = hn;
            y += hn * Cf[n];
        }
        y += xv * Dd;
        const float sz = zv / (1.f + __expf(-zv));
        y_bf16[rr * DI + d] = (__bf16)(y * sz);
    }
}

// ---------------------------------------------------------------------------
extern "C" void kernel_launch(void* const* d_in, const int* in_sizes, int n_in,
                              void* d_out, int out_size, void* d_ws, size_t ws_size,
                              hipStream_t stream) {
    const float* hs        = (const float*)d_in[0];  // (B,L,DM)
    const float* in_proj_w = (const float*)d_in[1];  // (2*DI, DM)
    const float* conv_w    = (const float*)d_in[2];  // (DI,1,4)
    const float* conv_b    = (const float*)d_in[3];  // (DI)
    const float* x_proj_w  = (const float*)d_in[4];  // (96, DI)
    const float* dt_proj_w = (const float*)d_in[5];  // (DI, DTR)
    const float* dt_proj_b = (const float*)d_in[6];  // (DI)
    const float* A_log     = (const float*)d_in[7];  // (DI, DS)
    const float* Dp        = (const float*)d_in[8];  // (DI)
    const float* out_proj_w= (const float*)d_in[9];  // (DM, DI)
    float* out = (float*)d_out;                      // (B,L,DM)

    char* ws = (char*)d_ws;
    size_t off = 0;
    auto alloc = [&](size_t bytes) -> char* {
        char* p = ws + off;
        off += (bytes + 255) & ~(size_t)255;
        return p;
    };
    __bf16* hs_bf        = (__bf16*)alloc((size_t)NTOK * DM * 2);
    __bf16* in_w_bf      = (__bf16*)alloc((size_t)2 * DI * DM * 2);
    __bf16* xp_w_bf      = (__bf16*)alloc((size_t)XPROJ_N * DI * 2);
    __bf16* dt_w_bf      = (__bf16*)alloc((size_t)DI * DTR * 2);
    __bf16* out_w_bf     = (__bf16*)alloc((size_t)DM * DI * 2);
    float*  xz           = (float*) alloc((size_t)NTOK * 2 * DI * 4);
    float*  xc_f32       = (float*) alloc((size_t)NTOK * DI * 4);
    __bf16* xc_bf        = (__bf16*)alloc((size_t)NTOK * DI * 2);
    float*  x_dbl        = (float*) alloc((size_t)NTOK * XPROJ_N * 4);
    __bf16* dt_low_bf    = (__bf16*)alloc((size_t)NTOK * DTR * 2);
    float*  dtmm         = (float*) alloc((size_t)NTOK * DI * 4);
    __bf16* y_bf         = (__bf16*)alloc((size_t)NTOK * DI * 2);
    (void)ws_size; (void)in_sizes; (void)n_in; (void)out_size;

    const int CT = 256;
    auto cblk = [](size_t n, int bs) { return (int)((n + bs - 1) / bs); };

    // 1) f32 -> bf16 staging
    cvt_f32_bf16<<<cblk((size_t)NTOK * DM, CT), CT, 0, stream>>>(hs, hs_bf, (size_t)NTOK * DM);
    cvt_f32_bf16<<<cblk((size_t)2 * DI * DM, CT), CT, 0, stream>>>(in_proj_w, in_w_bf, (size_t)2 * DI * DM);
    cvt_f32_bf16<<<cblk((size_t)XPROJ_N * DI, CT), CT, 0, stream>>>(x_proj_w, xp_w_bf, (size_t)XPROJ_N * DI);
    cvt_f32_bf16<<<cblk((size_t)DI * DTR, CT), CT, 0, stream>>>(dt_proj_w, dt_w_bf, (size_t)DI * DTR);
    cvt_f32_bf16<<<cblk((size_t)DM * DI, CT), CT, 0, stream>>>(out_proj_w, out_w_bf, (size_t)DM * DI);

    // 2) xz = hs @ in_proj_w^T   (16384 x 4096 x 1024), LDS/async staged
    gemm_bf16_wmma_lds<<<dim3((2 * DI) / 64, NTOK / 64), 128, 0, stream>>>(
        hs_bf, in_w_bf, xz, NTOK, 2 * DI, DM);

    // 3) depthwise causal conv + SiLU
    conv_silu<<<cblk((size_t)NTOK * DI, CT), CT, 0, stream>>>(xz, conv_w, conv_b, xc_f32, xc_bf);

    // 4) x_dbl = x @ x_proj_w^T  (16384 x 96 x 2048), direct (N=96)
    gemm_bf16_wmma<<<dim3(XPROJ_N / 32, NTOK / 32), 32, 0, stream>>>(
        xc_bf, xp_w_bf, x_dbl, NTOK, XPROJ_N, DI);

    // 5) dt = softplus(dt_low @ dt_proj_w^T + b)   (16384 x 2048 x 64)
    extract_dt_low<<<cblk((size_t)NTOK * DTR, CT), CT, 0, stream>>>(x_dbl, dt_low_bf, (size_t)NTOK * DTR);
    gemm_bf16_wmma_lds<<<dim3(DI / 64, NTOK / 64), 128, 0, stream>>>(
        dt_low_bf, dt_w_bf, dtmm, NTOK, DI, DTR);
    dt_softplus<<<cblk((size_t)NTOK * DI, CT), CT, 0, stream>>>(dtmm, dt_proj_b, (size_t)NTOK * DI);

    // 6) selective scan (+ x*D, * silu(z)), emits y in bf16
    selective_scan<<<(BATCH * DI) / CT, CT, 0, stream>>>(
        dtmm, xc_f32, xz, x_dbl, A_log, Dp, y_bf);

    // 7) out = y @ out_proj_w^T  (16384 x 1024 x 2048), LDS/async staged
    gemm_bf16_wmma_lds<<<dim3(DM / 64, NTOK / 64), 128, 0, stream>>>(
        y_bf, out_w_bf, out, NTOK, DM, DI);
}
